// DilateAttention_3865470566729
// MI455X (gfx1250) — matrix operand order repaired and checked by reading, not compiled, not run on hardware
//
#include <hip/hip_runtime.h>

typedef __attribute__((ext_vector_type(16))) _Float16 v16h;
typedef __attribute__((ext_vector_type(8)))  float    v8f;

#define HEAD_DIM 32
#define NH       12
#define HW       56
#define DIL      2
#define KSQ      9
#define TILE     16
#define WTILES   4                       // ceil(56/16)
#define SCALE    0.17677669529663687f    // 32^-0.5
#define WAVES_PER_BLOCK 8

__device__ __forceinline__ int reflect56(int t) {
    // ReflectionPad2d semantics: x[-1]->x[1], x[56]->x[54]
    t = (t < 0) ? -t : t;
    t = (t >= HW) ? (2 * (HW - 1) - t) : t;
    return t;
}

__global__ __launch_bounds__(256)
void dilate_attn_wmma_kernel(const float* __restrict__ q,
                             const float* __restrict__ k,
                             const float* __restrict__ v,
                             float* __restrict__ out,
                             int n_tiles) {
    // per-wave scratch: 16 pos x 32 ch floats (reused: scores then out-staging)
    __shared__ float lds[WAVES_PER_BLOCK][TILE * HEAD_DIM];

    const int lane = threadIdx.x & 31;
    const int wave = threadIdx.x >> 5;
    const int tile = blockIdx.x * WAVES_PER_BLOCK + wave;
    if (tile >= n_tiles) return;                    // wave-uniform

    const int wt   = tile & (WTILES - 1);
    const int h    = (tile >> 2) % HW;
    const int head = ((tile >> 2) / HW) % NH;
    const int b    = tile / (WTILES * HW * NH);
    const int w0   = wt * TILE;

    const int p    = lane & 15;                     // query index in tile
    const int half = lane >> 4;                     // channel half owned by lane
    const int wq   = w0 + p;
    const int wqc  = (wq < HW) ? wq : (HW - 1);     // clamped for loads

    const size_t plane = (size_t)HW * HW;           // 3136
    const size_t cbase = ((size_t)b * (NH * HEAD_DIM) + head * HEAD_DIM);

    // ---------------- A fragment: Q tile 16x32 -> f16 -----------------------
    // lane<16 : query M=lane, channels {0..7,16..23}; lane>=16: {8..15,24..31}
    v16h afrag;
    {
        const size_t qrow = cbase * plane + (size_t)h * HW + wqc;
        #pragma unroll
        for (int s = 0; s < 16; ++s) {
            const int c = ((s < 8) ? s : s + 8) + half * 8;
            afrag[s] = (_Float16)q[qrow + (size_t)c * plane];
        }
    }

    float* sc = &lds[wave][0];                      // scores sc[kk*16 + p]

    // ---------------- QK^T: one WMMA per kernel offset ----------------------
    #pragma unroll
    for (int kk = 0; kk < KSQ; ++kk) {
        const int kh = kk / 3, kw = kk % 3;
        const int hh = reflect56(h + DIL * (kh - 1));
        const int ww = reflect56(wqc + DIL * (kw - 1));  // key col for query N=p

        // B fragment: 32(K ch) x 16(N keys); lane<16 ch {0..7,16..23}, lane>=16 ch {8..15,24..31}
        v16h bfrag;
        const size_t krow = cbase * plane + (size_t)hh * HW + ww;
        #pragma unroll
        for (int s = 0; s < 16; ++s) {
            const int c = ((s < 8) ? s : s + 8) + half * 8;
            bfrag[s] = (_Float16)k[krow + (size_t)c * plane];
        }

        v8f acc = {};
        acc = __builtin_amdgcn_wmma_f32_16x16x32_f16(
            /*neg_a=*/false, afrag, /*neg_b=*/false, bfrag,
            /*c_mod=*/(short)0, acc, /*reuse_a=*/false, /*reuse_b=*/false);

        // diagonal S[p][p]: VGPR r @ lane r (p=r) and VGPR r @ lane 24+r (p=8+r)
        float d = 0.f;
        #pragma unroll
        for (int r = 0; r < 8; ++r)
            if (lane == r || lane == 24 + r) d = acc[r];
        if (lane < 8)        sc[kk * 16 + lane]        = d;
        else if (lane >= 24) sc[kk * 16 + (lane - 16)] = d;
    }

    // ---------------- softmax over the 9 window scores ----------------------
    float s9[KSQ];
    float m = -1e30f;
    #pragma unroll
    for (int kk = 0; kk < KSQ; ++kk) {
        s9[kk] = sc[kk * 16 + p] * SCALE;
        m = fmaxf(m, s9[kk]);
    }
    float sum = 0.f;
    #pragma unroll
    for (int kk = 0; kk < KSQ; ++kk) {
        s9[kk] = __expf(s9[kk] - m);
        sum += s9[kk];
    }
    const float inv = 1.0f / sum;

    // ---------------- AV: diag-scaled accumulation (VALU) -------------------
    float oacc[16];
    #pragma unroll
    for (int s = 0; s < 16; ++s) oacc[s] = 0.f;

    #pragma unroll
    for (int kk = 0; kk < KSQ; ++kk) {
        const int kh = kk / 3, kw = kk % 3;
        const int hh = reflect56(h + DIL * (kh - 1));
        const int ww = reflect56(wqc + DIL * (kw - 1));
        const float wgt = s9[kk] * inv;
        const size_t vrow = cbase * plane + (size_t)hh * HW + ww;
        #pragma unroll
        for (int s = 0; s < 16; ++s) {
            const int c = s + half * 16;
            oacc[s] = fmaf(wgt, v[vrow + (size_t)c * plane], oacc[s]);
        }
    }

    // ---------------- transpose via LDS, coalesced channels-last store ------
    float* ob = &lds[wave][0];                      // ob[p*32 + c]
    #pragma unroll
    for (int s = 0; s < 16; ++s)
        ob[p * HEAD_DIM + half * 16 + s] = oacc[s];

    // out: [B, H, W, NH*HEAD_DIM]; 32 consecutive floats per (pos, head)
    const size_t obase =
        (((size_t)b * HW + h) * HW) * (NH * HEAD_DIM) + head * HEAD_DIM;
    #pragma unroll
    for (int pp = 0; pp < TILE; ++pp) {
        const int wqq = w0 + pp;
        if (wqq < HW)
            out[obase + (size_t)wqq * (NH * HEAD_DIM) + lane] =
                ob[pp * HEAD_DIM + lane];
    }
}

extern "C" void kernel_launch(void* const* d_in, const int* in_sizes, int n_in,
                              void* d_out, int out_size, void* d_ws, size_t ws_size,
                              hipStream_t stream) {
    const float* q = (const float*)d_in[0];
    const float* k = (const float*)d_in[1];
    const float* v = (const float*)d_in[2];
    float* out = (float*)d_out;

    const int B = in_sizes[0] / (NH * HEAD_DIM * HW * HW);   // 16
    const int n_tiles = B * NH * HW * WTILES;                // 43008
    const int n_blocks = (n_tiles + WAVES_PER_BLOCK - 1) / WAVES_PER_BLOCK;

    hipLaunchKernelGGL(dilate_attn_wmma_kernel,
                       dim3(n_blocks), dim3(256), 0, stream,
                       q, k, v, out, n_tiles);
}
